// ImgGLULayer_36206574305377
// MI455X (gfx1250) — compile-verified, matrix operand order checked by reading, not run on hardware
//
#include <hip/hip_runtime.h>
#include <hip/hip_bf16.h>
#include <math.h>
#include <stdint.h>

typedef __bf16 bf16;
typedef __attribute__((ext_vector_type(8)))  bf16  v8bf;
typedef __attribute__((ext_vector_type(16))) bf16  v16bf;
typedef __attribute__((ext_vector_type(8)))  float v8f;

// ---------------------------------------------------------------------------
// CDNA5 async global->LDS copy (ASYNCcnt-tracked, no VGPR destination).
// lds: 32-bit LDS byte address (low 32 bits of a generic LDS pointer)
// gaddr: 64-bit global byte address
// ---------------------------------------------------------------------------
__device__ __forceinline__ void async_load_b128(uint32_t lds, uint64_t gaddr) {
    asm volatile("global_load_async_to_lds_b128 %0, %1, off"
                 :: "v"(lds), "v"(gaddr) : "memory");
}

template <int N>
__device__ __forceinline__ void s_wait_async() {
    if constexpr (N == 0)      asm volatile("s_wait_asynccnt 0x0" ::: "memory");
    else if constexpr (N == 6) asm volatile("s_wait_asynccnt 0x6" ::: "memory");
    else                       asm volatile("s_wait_asynccnt 0x8" ::: "memory");
}

__device__ __forceinline__ uint32_t lds_addr32(const void* p) {
    return (uint32_t)(uintptr_t)p;   // low 32 bits of generic LDS pointer = LDS offset
}

// ---------------------------------------------------------------------------
// fp32 -> bf16 conversion (weights, one-shot)
// ---------------------------------------------------------------------------
__global__ void cvt_bf16_kernel(const float* __restrict__ in, bf16* __restrict__ out, int n) {
    int i = blockIdx.x * blockDim.x + threadIdx.x;
    if (i < n) out[i] = (bf16)in[i];
}

// ---------------------------------------------------------------------------
// RMSNorm: one block (256 threads) per row
// ---------------------------------------------------------------------------
__global__ __launch_bounds__(256)
void rmsnorm_bf16_kernel(const float* __restrict__ x, const float* __restrict__ w,
                         bf16* __restrict__ y, int d) {
    const int row = blockIdx.x;
    const float* xr = x + (size_t)row * d;
    bf16* yr = y + (size_t)row * d;
    float ss = 0.f;
    for (int i = threadIdx.x; i < d; i += blockDim.x) { float v = xr[i]; ss += v * v; }
    for (int off = 16; off > 0; off >>= 1) ss += __shfl_xor(ss, off, 32);
    __shared__ float red[8];
    const int lane = threadIdx.x & 31, wid = threadIdx.x >> 5;
    if (lane == 0) red[wid] = ss;
    __syncthreads();
    float tot = 0.f;
    const int nw = blockDim.x >> 5;
    for (int i = 0; i < nw; ++i) tot += red[i];
    const float rinv = rsqrtf(tot / (float)d + 1e-6f);
    for (int i = threadIdx.x; i < d; i += blockDim.x)
        yr[i] = (bf16)(xr[i] * rinv * w[i]);
}

// ---------------------------------------------------------------------------
// Tiled WMMA bf16 GEMM with double-buffered async global->LDS pipeline.
//   C[M,N] = act(A[M,K] @ B^T[N,K] + bias) (+ addend)
//   BM=128, BN=64, BK=64; 256 threads = 8 waves (4 x 2), wave tile 32x32.
//   ACT: 0=none, 1=gelu(exact), 2=sigmoid. DUAL: two B matrices, epilogue
//   p = gelu(z0+b0) * (z1+b1). HAS_ADD: add addend[row*N+col] before store.
// ---------------------------------------------------------------------------
__device__ __forceinline__ float gelu_exact(float z) {
    return 0.5f * z * (1.f + erff(z * 0.70710678118f));
}

template <int ACT, typename OutT, bool DUAL, bool HAS_ADD>
__global__ __launch_bounds__(256)
void wmma_gemm_kernel(const bf16* __restrict__ A,
                      const bf16* __restrict__ B0, const bf16* __restrict__ B1,
                      const float* __restrict__ bias0, const float* __restrict__ bias1,
                      const float* __restrict__ addend,
                      OutT* __restrict__ Cout,
                      int M, int N, int K) {
    constexpr int BM = 128, BN = 64, BK = 64;
    constexpr int LDSS = 72;                       // 144B row stride: 16B aligned, 36-bank rotate
    constexpr int NSTAGE_LOADS = DUAL ? 8 : 6;     // async instrs issued per stage per lane
    __shared__ bf16 sA[2][BM * LDSS];
    __shared__ bf16 sB0[2][BN * LDSS];
    __shared__ bf16 sB1[DUAL ? 2 : 1][DUAL ? BN * LDSS : 1];

    const int tid   = threadIdx.x;
    const int blkN  = blockIdx.x * BN;
    const int blkM  = blockIdx.y * BM;
    const int wave  = tid >> 5;
    const int lane  = tid & 31;
    const int wm    = wave & 3;   // M direction (4)
    const int wn    = wave >> 2;  // N direction (2)
    const int m16   = lane & 15;
    const int khalf = lane >> 4;

    v8f acc0[2][2] = {};
    v8f acc1[2][2] = {};

    // per-lane staging geometry: 16B (8 bf16) per async load
    const int ld_row = tid >> 3;        // 0..31 within a 32-row chunk
    const int ld_col = (tid & 7) * 8;   // 0..56

    auto issue_stage = [&](int k0, int buf) {
        // A tile: 128 x 64 = 4 chunks of 32 rows
        for (int c = 0; c < 4; ++c) {
            const int row = c * 32 + ld_row;
            async_load_b128(lds_addr32(&sA[buf][row * LDSS + ld_col]),
                            (uint64_t)(uintptr_t)(A + (size_t)(blkM + row) * K + k0 + ld_col));
        }
        // B tile(s): 64 x 64 = 2 chunks of 32 rows
        for (int c = 0; c < 2; ++c) {
            const int row = c * 32 + ld_row;
            async_load_b128(lds_addr32(&sB0[buf][row * LDSS + ld_col]),
                            (uint64_t)(uintptr_t)(B0 + (size_t)(blkN + row) * K + k0 + ld_col));
            if constexpr (DUAL)
                async_load_b128(lds_addr32(&sB1[buf][row * LDSS + ld_col]),
                                (uint64_t)(uintptr_t)(B1 + (size_t)(blkN + row) * K + k0 + ld_col));
        }
    };

    const int nk = K / BK;
    issue_stage(0, 0);

    for (int kb = 0; kb < nk; ++kb) {
        if (kb + 1 < nk) {
            issue_stage((kb + 1) * BK, (kb + 1) & 1);  // prefetch next stage into other buffer
            s_wait_async<NSTAGE_LOADS>();              // stage kb landed; next stage in flight
        } else {
            s_wait_async<0>();
        }
        __syncthreads();                               // all waves' stage-kb data visible

        const bf16* cA  = &sA[kb & 1][0];
        const bf16* cB0 = &sB0[kb & 1][0];
        const bf16* cB1 = DUAL ? &sB1[kb & 1][0] : nullptr;

        for (int ks = 0; ks < 2; ++ks) {               // two K=32 sub-steps per stage
            const int kc = ks * 32;
            v16bf afrag[2], bfrag0[2], bfrag1[2];
            for (int tm = 0; tm < 2; ++tm) {
                const int r = wm * 32 + tm * 16 + m16;
                v8bf lo = *(const v8bf*)(cA + r * LDSS + kc + khalf * 8);
                v8bf hi = *(const v8bf*)(cA + r * LDSS + kc + 16 + khalf * 8);
                afrag[tm] = __builtin_shufflevector(lo, hi, 0,1,2,3,4,5,6,7,8,9,10,11,12,13,14,15);
            }
            for (int tn = 0; tn < 2; ++tn) {
                const int r = wn * 32 + tn * 16 + m16;
                v8bf lo = *(const v8bf*)(cB0 + r * LDSS + kc + khalf * 8);
                v8bf hi = *(const v8bf*)(cB0 + r * LDSS + kc + 16 + khalf * 8);
                bfrag0[tn] = __builtin_shufflevector(lo, hi, 0,1,2,3,4,5,6,7,8,9,10,11,12,13,14,15);
                if constexpr (DUAL) {
                    v8bf lo1 = *(const v8bf*)(cB1 + r * LDSS + kc + khalf * 8);
                    v8bf hi1 = *(const v8bf*)(cB1 + r * LDSS + kc + 16 + khalf * 8);
                    bfrag1[tn] = __builtin_shufflevector(lo1, hi1, 0,1,2,3,4,5,6,7,8,9,10,11,12,13,14,15);
                }
            }
            for (int tm = 0; tm < 2; ++tm)
                for (int tn = 0; tn < 2; ++tn) {
                    acc0[tm][tn] = __builtin_amdgcn_wmma_f32_16x16x32_bf16(
                        false, afrag[tm], false, bfrag0[tn], (short)0, acc0[tm][tn], false, false);
                    if constexpr (DUAL)
                        acc1[tm][tn] = __builtin_amdgcn_wmma_f32_16x16x32_bf16(
                            false, afrag[tm], false, bfrag1[tn], (short)0, acc1[tm][tn], false, false);
                }
        }
        __syncthreads();   // everyone done reading buf[kb&1] before it is overwritten
    }

    // epilogue: C layout lane l: col = l&15, row = 8*(l>>4) + vgpr
    for (int tm = 0; tm < 2; ++tm)
        for (int tn = 0; tn < 2; ++tn) {
            const int col  = blkN + wn * 32 + tn * 16 + m16;
            const int row0 = blkM + wm * 32 + tm * 16 + khalf * 8;
            const float b0v = bias0[col];
            float b1v = 0.f;
            if constexpr (DUAL) b1v = bias1[col];
            v8f c0 = acc0[tm][tn];
            v8f c1 = acc1[tm][tn];
            for (int v = 0; v < 8; ++v) {
                const size_t oi = (size_t)(row0 + v) * N + col;
                const float z = c0[v] + b0v;
                float r;
                if constexpr (DUAL) {
                    r = gelu_exact(z) * (c1[v] + b1v);
                } else if constexpr (ACT == 1) {
                    r = gelu_exact(z);
                } else if constexpr (ACT == 2) {
                    r = 1.f / (1.f + expf(-z));
                } else {
                    r = z;
                }
                if constexpr (HAS_ADD) r += addend[oi];
                Cout[oi] = (OutT)r;
            }
        }
}

// ---------------------------------------------------------------------------
// fused elementwise + linear recurrence + gate + residual:
//   a = exp(-8*softplus(alpha)*r); s = sqrt(1-a^2)*hs*i
//   o[t] = a[t]*o[t-1] + s[t];  hs2 = o*gate + residual
// one thread per (b,d); coalesced along d; prefetch 8 steps ahead.
// ---------------------------------------------------------------------------
__global__ __launch_bounds__(256)
void scan_kernel(const float* __restrict__ alpha,
                 const bf16* __restrict__ rbuf, const bf16* __restrict__ ibuf,
                 const float* __restrict__ hsbuf, const bf16* __restrict__ gatebuf,
                 const float* __restrict__ resid, float* __restrict__ hs2,
                 int Bn, int Ln, int Dn) {
    const int idx = blockIdx.x * blockDim.x + threadIdx.x;
    if (idx >= Bn * Dn) return;
    const int b = idx / Dn, d = idx % Dn;
    const float sp = log1pf(expf(alpha[d]));
    const float c8 = -8.f * sp;
    float o = 0.f;
    size_t base = (size_t)b * Ln * Dn + d;
    for (int t = 0; t < Ln; ++t, base += Dn) {
        if ((t & 7) == 0 && t + 8 < Ln) {
            const size_t pb = base + (size_t)8 * Dn;
            __builtin_prefetch(hsbuf + pb, 0, 1);
            __builtin_prefetch(rbuf + pb, 0, 1);
            __builtin_prefetch(ibuf + pb, 0, 1);
            __builtin_prefetch(gatebuf + pb, 0, 1);
            __builtin_prefetch(resid + pb, 0, 1);
        }
        const float rv = (float)rbuf[base];
        const float iv = (float)ibuf[base];
        const float hv = hsbuf[base];
        const float a  = expf(c8 * rv);
        const float s  = sqrtf(fmaxf(0.f, 1.f - a * a)) * hv * iv;
        o = a * o + s;
        hs2[base] = o * (float)gatebuf[base] + resid[base];
    }
}

// ---------------------------------------------------------------------------
extern "C" void kernel_launch(void* const* d_in, const int* in_sizes, int n_in,
                              void* d_out, int out_size, void* d_ws, size_t ws_size,
                              hipStream_t stream) {
    (void)in_sizes; (void)n_in; (void)out_size; (void)ws_size;
    constexpr int Dm = 1024, Bm = 16, Lm = 2048, D3 = 3 * Dm;
    constexpr int Ntok = Bm * Lm;  // 32768

    const float* hidden    = (const float*)d_in[0];
    const float* alpha     = (const float*)d_in[1];
    const float* fc_w      = (const float*)d_in[2];
    const float* fc_b      = (const float*)d_in[3];
    const float* fc_r_w    = (const float*)d_in[4];
    const float* fc_r_b    = (const float*)d_in[5];
    const float* fc_i_w    = (const float*)d_in[6];
    const float* fc_i_b    = (const float*)d_in[7];
    const float* fc_gate_w = (const float*)d_in[8];
    const float* fc_gate_b = (const float*)d_in[9];
    const float* norm_w    = (const float*)d_in[10];
    const float* norm2_w   = (const float*)d_in[11];
    const float* mlp_gate_w = (const float*)d_in[12];
    const float* mlp_gate_b = (const float*)d_in[13];
    const float* mlp_fc_w   = (const float*)d_in[14];
    const float* mlp_fc_b   = (const float*)d_in[15];
    const float* out_w      = (const float*)d_in[16];
    const float* out_b      = (const float*)d_in[17];
    float* outp = (float*)d_out;

    char* ws = (char*)d_ws;
    size_t off = 0;
    auto alloc = [&](size_t bytes) -> void* {
        void* p = ws + off;
        off = (off + bytes + 255) & ~(size_t)255;
        return p;
    };

    bf16* h_bf   = (bf16*)alloc((size_t)Ntok * Dm * 2);
    bf16* wg_bf  = (bf16*)alloc((size_t)Dm * Dm * 2);
    bf16* wfc_bf = (bf16*)alloc((size_t)Dm * Dm * 2);
    bf16* wr_bf  = (bf16*)alloc((size_t)Dm * Dm * 2);
    bf16* wi_bf  = (bf16*)alloc((size_t)Dm * Dm * 2);
    bf16* wmg_bf = (bf16*)alloc((size_t)D3 * Dm * 2);
    bf16* wmf_bf = (bf16*)alloc((size_t)D3 * Dm * 2);
    bf16* wo_bf  = (bf16*)alloc((size_t)Dm * D3 * 2);
    bf16* gate_bf = (bf16*)alloc((size_t)Ntok * Dm * 2);
    float* hs_f   = (float*)alloc((size_t)Ntok * Dm * 4);
    bf16* r_bf    = (bf16*)alloc((size_t)Ntok * Dm * 2);
    bf16* i_bf    = (bf16*)alloc((size_t)Ntok * Dm * 2);
    float* hs2_f  = (float*)alloc((size_t)Ntok * Dm * 4);
    bf16* h2_bf   = (bf16*)alloc((size_t)Ntok * Dm * 2);
    bf16* p_bf    = (bf16*)alloc((size_t)Ntok * D3 * 2);

    // 1. weights -> bf16
    const int nDD = Dm * Dm, n3D = D3 * Dm;
    cvt_bf16_kernel<<<(nDD + 255) / 256, 256, 0, stream>>>(fc_gate_w, wg_bf, nDD);
    cvt_bf16_kernel<<<(nDD + 255) / 256, 256, 0, stream>>>(fc_w, wfc_bf, nDD);
    cvt_bf16_kernel<<<(nDD + 255) / 256, 256, 0, stream>>>(fc_r_w, wr_bf, nDD);
    cvt_bf16_kernel<<<(nDD + 255) / 256, 256, 0, stream>>>(fc_i_w, wi_bf, nDD);
    cvt_bf16_kernel<<<(n3D + 255) / 256, 256, 0, stream>>>(mlp_gate_w, wmg_bf, n3D);
    cvt_bf16_kernel<<<(n3D + 255) / 256, 256, 0, stream>>>(mlp_fc_w, wmf_bf, n3D);
    cvt_bf16_kernel<<<(n3D + 255) / 256, 256, 0, stream>>>(out_w, wo_bf, n3D);

    // 2. rmsnorm -> h
    rmsnorm_bf16_kernel<<<Ntok, 256, 0, stream>>>(hidden, norm_w, h_bf, Dm);

    // 3. four D x D GEMMs with fused activations
    dim3 blk(256);
    dim3 g1(Dm / 64, Ntok / 128);
    wmma_gemm_kernel<1, bf16, false, false><<<g1, blk, 0, stream>>>(
        h_bf, wg_bf, nullptr, fc_gate_b, nullptr, nullptr, gate_bf, Ntok, Dm, Dm);
    wmma_gemm_kernel<0, float, false, false><<<g1, blk, 0, stream>>>(
        h_bf, wfc_bf, nullptr, fc_b, nullptr, nullptr, hs_f, Ntok, Dm, Dm);
    wmma_gemm_kernel<2, bf16, false, false><<<g1, blk, 0, stream>>>(
        h_bf, wr_bf, nullptr, fc_r_b, nullptr, nullptr, r_bf, Ntok, Dm, Dm);
    wmma_gemm_kernel<2, bf16, false, false><<<g1, blk, 0, stream>>>(
        h_bf, wi_bf, nullptr, fc_i_b, nullptr, nullptr, i_bf, Ntok, Dm, Dm);

    // 4. recurrence + gate + residual -> hs2
    scan_kernel<<<(Bm * Dm + 255) / 256, 256, 0, stream>>>(
        alpha, r_bf, i_bf, hs_f, gate_bf, hidden, hs2_f, Bm, Lm, Dm);

    // 5. rmsnorm2 -> h2
    rmsnorm_bf16_kernel<<<Ntok, 256, 0, stream>>>(hs2_f, norm2_w, h2_bf, Dm);

    // 6. dual GEMM (mlp_gate & mlp_fc share A), epilogue p = gelu(zg)*(zf)
    dim3 g2(D3 / 64, Ntok / 128);
    wmma_gemm_kernel<0, bf16, true, false><<<g2, blk, 0, stream>>>(
        h2_bf, wmg_bf, wmf_bf, mlp_gate_b, mlp_fc_b, nullptr, p_bf, Ntok, D3, Dm);

    // 7. final GEMM + bias + hs2 residual -> out
    wmma_gemm_kernel<0, float, false, true><<<g1, blk, 0, stream>>>(
        p_bf, wo_bf, nullptr, out_b, nullptr, hs2_f, outp, Ntok, Dm, D3);
}
